// NeuralMemoryMLP_69587060130333
// MI455X (gfx1250) — matrix operand order repaired and checked by reading, not compiled, not run on hardware
//
#include <hip/hip_runtime.h>
#include <stdint.h>

// ---------------------------------------------------------------------------
// NeuralMemoryMLP for MI455X (gfx1250): all GEMMs via v_wmma_f32_16x16x32_bf16,
// double-buffered LDS staging via GLOBAL_LOAD_ASYNC_TO_LDS_B128 (ASYNCcnt)
// overlapped with WMMA compute.
// ---------------------------------------------------------------------------

typedef __attribute__((ext_vector_type(16))) __bf16 v16bf;
typedef __attribute__((ext_vector_type(8)))  float  v8f;

#define INNER_LR 0.01f

__device__ __forceinline__ uint16_t f2bf(float f) {
    union { float f; uint32_t u; } v; v.f = f;
    uint32_t r = v.u + 0x7FFFu + ((v.u >> 16) & 1u);   // round-to-nearest-even
    return (uint16_t)(r >> 16);
}
__device__ __forceinline__ float bf2f(uint16_t h) {
    union { uint32_t u; float f; } v; v.u = ((uint32_t)h) << 16;
    return v.f;
}

// Async 16-byte global->LDS copy (gfx1250 GLOBAL_LOAD_ASYNC_TO_LDS_B128).
// ldsOff = low 32 bits of generic pointer to __shared__ (== LDS byte address).
__device__ __forceinline__ void async_g2l_b128(uint32_t ldsOff, const void* gptr) {
    asm volatile("global_load_async_to_lds_b128 %0, %1, off"
                 :: "v"(ldsOff), "v"(gptr)
                 : "memory");
}
__device__ __forceinline__ void wait_asynccnt0() {
    asm volatile("s_wait_asynccnt 0" ::: "memory");
}

// ---------------------------------------------------------------------------
// Tiled WMMA GEMM, double-buffered async pipeline.
//   TA=false : A is [M,K] row-major          (frag elem A[m*K+k])
//   TA=true  : A is [K,M] row-major (A^T op) (frag elem A[k*M+m])
//   TB=true  : B is [N,K] row-major (B^T op) (frag elem B[n*K+k])
//   TB=false : B is [K,N] row-major          (frag elem B[k*N+n])
// EPI: 0 = outF = acc + bias
//      1 = t=acc+bias; outH=bf16(silu(t)); outAux=bf16(dsilu(t))
//      2 = t=acc+bias; outH=bf16((t - vres)*scale)
//      3 = outH = bf16(acc * bf2f(aux))
// Block tile 128x128, K-step 32. 256 threads = 8 waves (4 row x 2 col),
// wave tile 32x64 = 2x4 WMMA 16x16 tiles.
// ---------------------------------------------------------------------------
template<bool TA, bool TB, int EPI>
__global__ __launch_bounds__(256)
void gemm_bf16_wmma(const uint16_t* __restrict__ A,
                    const uint16_t* __restrict__ B,
                    const float*    __restrict__ bias,
                    const float*    __restrict__ vres,
                    const uint16_t* __restrict__ aux,
                    float scale,
                    float*    __restrict__ outF,
                    uint16_t* __restrict__ outH,
                    uint16_t* __restrict__ outAux,
                    int M, int N, int K)
{
    constexpr int BM = 128, BN = 128, BK = 32, LDST = 40; // padded LDS stride
    __shared__ uint16_t As[2][BM * LDST];
    __shared__ uint16_t Bs[2][BN * LDST];

    const int tid  = threadIdx.x;
    const int lane = tid & 31;
    const int wave = tid >> 5;
    const int waveRow = wave & 3;   // 0..3 -> 32 rows each
    const int waveCol = wave >> 2;  // 0..1 -> 64 cols each
    const int bm0 = blockIdx.y * BM;
    const int bn0 = blockIdx.x * BN;

    v8f acc[2][4];
    {
        v8f zero = {0.f,0.f,0.f,0.f,0.f,0.f,0.f,0.f};
        for (int i = 0; i < 2; ++i)
            for (int j = 0; j < 4; ++j) acc[i][j] = zero;
    }

    union Frag { uint4 q[2]; v16bf v; };
    union Pack { uint4 q; uint16_t h[16]; };

    // Per-thread staging coordinates (contiguous-copy paths).
    const int crow = tid >> 1, cchunk = tid & 1;
    // Per-thread staging coordinates (scatter paths).
    const int skk = tid & 31, schunk = tid >> 5;

    const uint32_t ldsA[2] = {
        (uint32_t)(uintptr_t)&As[0][crow * LDST + cchunk * 16],
        (uint32_t)(uintptr_t)&As[1][crow * LDST + cchunk * 16] };
    const uint32_t ldsB[2] = {
        (uint32_t)(uintptr_t)&Bs[0][crow * LDST + cchunk * 16],
        (uint32_t)(uintptr_t)&Bs[1][crow * LDST + cchunk * 16] };

    // Stage one 128x32 (A) + 128x32 (B) K-tile into LDS buffer `buf`.
    auto stage = [&](int buf, int kk0) {
        if (!TA) {
            async_g2l_b128(ldsA[buf], A + (size_t)(bm0 + crow) * K + kk0 + cchunk * 16);
        } else {
            Pack u;
            u.q = *(const uint4*)(A + (size_t)(kk0 + skk) * M + bm0 + schunk * 16);
            #pragma unroll
            for (int j = 0; j < 16; ++j) As[buf][(schunk * 16 + j) * LDST + skk] = u.h[j];
        }
        if (TB) {
            async_g2l_b128(ldsB[buf], B + (size_t)(bn0 + crow) * K + kk0 + cchunk * 16);
        } else {
            Pack u;
            u.q = *(const uint4*)(B + (size_t)(kk0 + skk) * N + bn0 + schunk * 16);
            #pragma unroll
            for (int j = 0; j < 16; ++j) Bs[buf][(schunk * 16 + j) * LDST + skk] = u.h[j];
        }
    };

    // ---- pipeline prologue: stage tile 0 ----
    stage(0, 0);
    wait_asynccnt0();
    __syncthreads();

    int ib = 0;
    for (int k0 = 0; k0 < K; k0 += BK, ib ^= 1) {
        // issue async copy of the NEXT K-tile into the other buffer
        if (k0 + BK < K) stage(ib ^ 1, k0 + BK);
        // prefetch hint two tiles ahead (global_prefetch_b8)
        if (k0 + 2 * BK < K) {
            const uint16_t* pa = TA ? (A + (size_t)(k0 + 2 * BK) * M + bm0)
                                    : (A + (size_t)bm0 * K + k0 + 2 * BK);
            const uint16_t* pb = TB ? (B + (size_t)bn0 * K + k0 + 2 * BK)
                                    : (B + (size_t)(k0 + 2 * BK) * N + bn0);
            __builtin_prefetch(pa, 0, 3);
            __builtin_prefetch(pb, 0, 3);
        }

        // ---- compute current tile from buffer ib ----
        const uint16_t* Ac = &As[ib][0];
        const uint16_t* Bc = &Bs[ib][0];
        // A 16x32: lane<16 holds row M=lane, K {0..7,16..23}; lane>=16 row M=lane-16, K {8..15,24..31}
        Frag aF[2];
        const int ksel = (lane >> 4) * 8;
        #pragma unroll
        for (int tm = 0; tm < 2; ++tm) {
            int r = waveRow * 32 + tm * 16 + (lane & 15);
            aF[tm].q[0] = *(const uint4*)(&Ac[r * LDST + ksel]);
            aF[tm].q[1] = *(const uint4*)(&Ac[r * LDST + ksel + 16]);
        }
        // B 32x16: lane<16 holds col N=lane, K {0..7,16..23}; lane>=16 col N=lane-16, K {8..15,24..31}
        Frag bF[4];
        const int koff = (lane >> 4) * 8;
        #pragma unroll
        for (int tn = 0; tn < 4; ++tn) {
            int c = waveCol * 64 + tn * 16 + (lane & 15);
            bF[tn].q[0] = *(const uint4*)(&Bc[c * LDST + koff]);
            bF[tn].q[1] = *(const uint4*)(&Bc[c * LDST + koff + 16]);
        }
        #pragma unroll
        for (int tm = 0; tm < 2; ++tm)
            #pragma unroll
            for (int tn = 0; tn < 4; ++tn)
                acc[tm][tn] = __builtin_amdgcn_wmma_f32_16x16x32_bf16(
                    false, aF[tm].v, false, bF[tn].v,
                    (short)0, acc[tm][tn], false, false);

        // next tile's async writes complete (this wave), then all waves align;
        // also guarantees this tile's ds_loads are consumed before overwrite.
        wait_asynccnt0();
        __syncthreads();
    }

    // ---- epilogue: C 16x16 f32 layout: VGPR i -> row = base + 8*(lane>>4) + i, col = base + (lane&15)
    #pragma unroll
    for (int tm = 0; tm < 2; ++tm) {
        #pragma unroll
        for (int tn = 0; tn < 4; ++tn) {
            int rowBase = bm0 + waveRow * 32 + tm * 16 + ((lane >> 4) * 8);
            int col     = bn0 + waveCol * 64 + tn * 16 + (lane & 15);
            float bv = 0.f;
            if constexpr (EPI != 3) { if (bias) bv = bias[col]; }
            #pragma unroll
            for (int i = 0; i < 8; ++i) {
                int row = rowBase + i;
                size_t idx = (size_t)row * N + col;
                float t = acc[tm][tn][i] + bv;
                if constexpr (EPI == 0) {
                    outF[idx] = t;
                } else if constexpr (EPI == 1) {
                    float s = 1.0f / (1.0f + __expf(-t));
                    float a = t * s;
                    float d = s * (1.0f + t * (1.0f - s));
                    outH[idx]   = f2bf(a);
                    outAux[idx] = f2bf(d);
                } else if constexpr (EPI == 2) {
                    outH[idx] = f2bf((t - vres[idx]) * scale);
                } else {
                    outH[idx] = f2bf(t * bf2f(aux[idx]));
                }
            }
        }
    }
}

// ---------------------------------------------------------------------------
// Elementwise / reduction helpers
// ---------------------------------------------------------------------------
__global__ __launch_bounds__(256)
void cast_f32_bf16(const float* __restrict__ in, uint16_t* __restrict__ out, int n) {
    int i = (blockIdx.x * 256 + threadIdx.x) * 4;
    if (i + 3 < n) {
        float4 f = *(const float4*)(in + i);
        union { uint16_t h[4]; uint2 u; } pk;
        pk.h[0] = f2bf(f.x); pk.h[1] = f2bf(f.y);
        pk.h[2] = f2bf(f.z); pk.h[3] = f2bf(f.w);
        *(uint2*)(out + i) = pk.u;
    }
}

__global__ __launch_bounds__(256)
void colsum_f32(const float* __restrict__ X, float* __restrict__ out,
                int S, int D, float scale) {
    int c = blockIdx.x * 256 + threadIdx.x;
    float acc = 0.f;
    for (int s = 0; s < S; ++s) acc += X[(size_t)s * D + c];
    out[c] = acc * scale;
}

__global__ __launch_bounds__(256)
void colsum_bf16(const uint16_t* __restrict__ X, float* __restrict__ out,
                 int S, int D, float scale) {
    int c = blockIdx.x * 256 + threadIdx.x;
    float acc = 0.f;
    for (int s = 0; s < S; ++s) acc += bf2f(X[(size_t)s * D + c]);
    out[c] = acc * scale;
}

__global__ __launch_bounds__(256)
void alpha_sigmoid(const float* __restrict__ meanx, const float* __restrict__ aw,
                   const float* __restrict__ ab, float* __restrict__ alphaOut, int D) {
    __shared__ float red[256];
    float p = 0.f;
    for (int j = threadIdx.x; j < D; j += 256) p += meanx[j] * aw[j];
    red[threadIdx.x] = p;
    __syncthreads();
    for (int s = 128; s > 0; s >>= 1) {
        if (threadIdx.x < s) red[threadIdx.x] += red[threadIdx.x + s];
        __syncthreads();
    }
    if (threadIdx.x == 0)
        alphaOut[0] = 1.0f / (1.0f + __expf(-(red[0] + ab[0])));
}

__global__ __launch_bounds__(256)
void rownorm_bf16(const float* __restrict__ X, uint16_t* __restrict__ out, int D) {
    __shared__ float red[256];
    int row = blockIdx.x;
    const float* xr = X + (size_t)row * D;
    float ss = 0.f;
    for (int j = threadIdx.x; j < D; j += 256) { float v = xr[j]; ss += v * v; }
    red[threadIdx.x] = ss;
    __syncthreads();
    for (int s = 128; s > 0; s >>= 1) {
        if (threadIdx.x < s) red[threadIdx.x] += red[threadIdx.x + s];
        __syncthreads();
    }
    float inv = 1.0f / fmaxf(sqrtf(red[0]), 1e-12f);
    uint16_t* orow = out + (size_t)row * D;
    for (int j = threadIdx.x; j < D; j += 256) orow[j] = f2bf(xr[j] * inv);
}

__global__ __launch_bounds__(256)
void fast_weight(const float* __restrict__ W, const float* __restrict__ G,
                 const float* __restrict__ alpha, uint16_t* __restrict__ out, int n) {
    int i = blockIdx.x * 256 + threadIdx.x;
    if (i < n) out[i] = f2bf((1.0f - alpha[0]) * W[i] - INNER_LR * G[i]);
}

__global__ __launch_bounds__(256)
void fast_bias(const float* __restrict__ b, const float* __restrict__ g,
               const float* __restrict__ alpha, float* __restrict__ out, int n) {
    int i = blockIdx.x * 256 + threadIdx.x;
    if (i < n) out[i] = (1.0f - alpha[0]) * b[i] - INNER_LR * g[i];
}

// ---------------------------------------------------------------------------
extern "C" void kernel_launch(void* const* d_in, const int* in_sizes, int n_in,
                              void* d_out, int out_size, void* d_ws, size_t ws_size,
                              hipStream_t stream)
{
    (void)in_sizes; (void)n_in; (void)out_size; (void)ws_size;

    const float* x  = (const float*)d_in[0];
    const float* Wq = (const float*)d_in[1];
    const float* Wk = (const float*)d_in[2];
    const float* Wv = (const float*)d_in[3];
    const float* aw = (const float*)d_in[4];
    const float* ab = (const float*)d_in[5];
    const float* W1 = (const float*)d_in[6];
    const float* b1 = (const float*)d_in[7];
    const float* W2 = (const float*)d_in[8];
    const float* b2 = (const float*)d_in[9];

    const int D = 2048, S = 8192;
    const size_t SD = (size_t)S * D, DD = (size_t)D * D;

    char* cur = (char*)d_ws;
    auto alloc = [&](size_t bytes) -> void* {
        void* p = (void*)cur;
        cur += (bytes + 255) & ~(size_t)255;
        return p;
    };
    uint16_t* xb   = (uint16_t*)alloc(SD * 2);
    uint16_t* wqb  = (uint16_t*)alloc(DD * 2);
    uint16_t* wkb  = (uint16_t*)alloc(DD * 2);
    uint16_t* wvb  = (uint16_t*)alloc(DD * 2);
    uint16_t* w1b  = (uint16_t*)alloc(DD * 2);
    uint16_t* w2b  = (uint16_t*)alloc(DD * 2);
    uint16_t* kb   = (uint16_t*)alloc(SD * 2);
    uint16_t* qb   = (uint16_t*)alloc(SD * 2);
    uint16_t* abuf = (uint16_t*)alloc(SD * 2);  // silu(h); reused for a2
    uint16_t* dbuf = (uint16_t*)alloc(SD * 2);  // dsilu(h); scratch in last MLP
    uint16_t* gbuf = (uint16_t*)alloc(SD * 2);  // dL/dy
    uint16_t* dhb  = (uint16_t*)alloc(SD * 2);  // dL/dh
    uint16_t* fw1b = (uint16_t*)alloc(DD * 2);
    uint16_t* fw2b = (uint16_t*)alloc(DD * 2);
    float* vF    = (float*)alloc(SD * 4);
    float* tmpF  = (float*)alloc(SD * 4);       // Kraw then Qraw
    float* gW1   = (float*)alloc(DD * 4);
    float* gW2   = (float*)alloc(DD * 4);
    float* meanx = (float*)alloc(D * 4);
    float* alphB = (float*)alloc(256);
    float* gb1   = (float*)alloc(D * 4);
    float* gb2   = (float*)alloc(D * 4);
    float* fb1   = (float*)alloc(D * 4);
    float* fb2   = (float*)alloc(D * 4);
    float* outp  = (float*)d_out;

    dim3 blk(256);
    dim3 gSD(D / 128, S / 128);  // [S,D] outputs
    dim3 gDD(D / 128, D / 128);  // [D,D] outputs
    const float gscale = 2.0f / (float)(SD);

    // 0) casts
    cast_f32_bf16<<<(int)(SD / 1024), blk, 0, stream>>>(x,  xb,  (int)SD);
    cast_f32_bf16<<<(int)(DD / 1024), blk, 0, stream>>>(Wq, wqb, (int)DD);
    cast_f32_bf16<<<(int)(DD / 1024), blk, 0, stream>>>(Wk, wkb, (int)DD);
    cast_f32_bf16<<<(int)(DD / 1024), blk, 0, stream>>>(Wv, wvb, (int)DD);
    cast_f32_bf16<<<(int)(DD / 1024), blk, 0, stream>>>(W1, w1b, (int)DD);
    cast_f32_bf16<<<(int)(DD / 1024), blk, 0, stream>>>(W2, w2b, (int)DD);

    // alpha = sigmoid(mean(x,S) . alpha_w + alpha_b)
    colsum_f32<<<D / 256, blk, 0, stream>>>(x, meanx, S, D, 1.0f / (float)S);
    alpha_sigmoid<<<1, blk, 0, stream>>>(meanx, aw, ab, alphB, D);

    // 1) k = l2norm(x @ Wk^T); q = l2norm(x @ Wq^T); v = x @ Wv^T
    gemm_bf16_wmma<false,true,0><<<gSD, blk, 0, stream>>>(xb, wkb, nullptr, nullptr, nullptr, 0.f, tmpF, nullptr, nullptr, S, D, D);
    rownorm_bf16<<<S, blk, 0, stream>>>(tmpF, kb, D);
    gemm_bf16_wmma<false,true,0><<<gSD, blk, 0, stream>>>(xb, wqb, nullptr, nullptr, nullptr, 0.f, tmpF, nullptr, nullptr, S, D, D);
    rownorm_bf16<<<S, blk, 0, stream>>>(tmpF, qb, D);
    gemm_bf16_wmma<false,true,0><<<gSD, blk, 0, stream>>>(xb, wvb, nullptr, nullptr, nullptr, 0.f, vF, nullptr, nullptr, S, D, D);

    // 2) forward MLP on k: a = silu(k@W1^T + b1), d = dsilu
    gemm_bf16_wmma<false,true,1><<<gSD, blk, 0, stream>>>(kb, w1b, b1, nullptr, nullptr, 0.f, nullptr, abuf, dbuf, S, D, D);
    // 3) G = (2/SD) * (a@W2^T + b2 - v)
    gemm_bf16_wmma<false,true,2><<<gSD, blk, 0, stream>>>(abuf, w2b, b2, vF, nullptr, gscale, nullptr, gbuf, nullptr, S, D, D);
    colsum_bf16<<<D / 256, blk, 0, stream>>>(gbuf, gb2, S, D, 1.0f);           // grad_b2
    // 4) dH = (G @ W2) * dsilu
    gemm_bf16_wmma<false,false,3><<<gSD, blk, 0, stream>>>(gbuf, w2b, nullptr, nullptr, dbuf, 0.f, nullptr, dhb, nullptr, S, D, D);
    colsum_bf16<<<D / 256, blk, 0, stream>>>(dhb, gb1, S, D, 1.0f);            // grad_b1
    // 5) grad_W2 = G^T @ a ; grad_W1 = dH^T @ k   (TN, K = S)
    gemm_bf16_wmma<true,false,0><<<gDD, blk, 0, stream>>>(gbuf, abuf, nullptr, nullptr, nullptr, 0.f, gW2, nullptr, nullptr, D, D, S);
    gemm_bf16_wmma<true,false,0><<<gDD, blk, 0, stream>>>(dhb, kb, nullptr, nullptr, nullptr, 0.f, gW1, nullptr, nullptr, D, D, S);

    // 6) fast weights: (1-alpha)*p - lr*g
    fast_weight<<<(int)(DD / 256), blk, 0, stream>>>(W1, gW1, alphB, fw1b, (int)DD);
    fast_weight<<<(int)(DD / 256), blk, 0, stream>>>(W2, gW2, alphB, fw2b, (int)DD);
    fast_bias<<<D / 256, blk, 0, stream>>>(b1, gb1, alphB, fb1, D);
    fast_bias<<<D / 256, blk, 0, stream>>>(b2, gb2, alphB, fb2, D);

    // 7) retrieve: out = silu(q@fW1^T + fb1) @ fW2^T + fb2
    gemm_bf16_wmma<false,true,1><<<gSD, blk, 0, stream>>>(qb, fw1b, fb1, nullptr, nullptr, 0.f, nullptr, abuf, dbuf, S, D, D);
    gemm_bf16_wmma<false,true,0><<<gSD, blk, 0, stream>>>(abuf, fw2b, fb2, nullptr, nullptr, 0.f, outp, nullptr, nullptr, S, D, D);
}
